// GAT_Decoder_44461501448711
// MI455X (gfx1250) — compile-verified, matrix operand order checked by reading, not run on hardware
//
#include <hip/hip_runtime.h>
#include <hip/hip_bf16.h>
#include <math.h>

#define NEGV (-1000000000.0f)

typedef __attribute__((ext_vector_type(2)))  float    v2f;
typedef __attribute__((ext_vector_type(8)))  float    v8f;
typedef __attribute__((ext_vector_type(16))) _Float16 v16h;

// ---------------------------------------------------------------------------
// Small dense helpers (tiny matrices, negligible cost)
// ---------------------------------------------------------------------------

// Wct[j,k] = sum_h Wo[h,j] * Wk2[h,k]   (so that K2o = E @ Wk2^T @ Wo = NT(E, Wct))
__global__ void k_wct(const float* __restrict__ Wo, const float* __restrict__ Wk2,
                      float* __restrict__ Wct, int H) {
    int j = blockIdx.x, k = threadIdx.x;
    float s = 0.f;
    for (int h = 0; h < H; ++h) s += Wo[h * H + j] * Wk2[h * H + k];
    Wct[j * H + k] = s;
}

// Wqf[h,j] = sum_i Wq[h,i] * fc_w[i,j],  j in [0, H]  (H+1 columns)
__global__ void k_wqf(const float* __restrict__ Wq, const float* __restrict__ fc_w,
                      float* __restrict__ Wqf, int H) {
    int h = blockIdx.x, j = threadIdx.x;
    if (j <= H) {
        float s = 0.f;
        for (int i = 0; i < H; ++i) s += Wq[h * H + i] * fc_w[i * (H + 1) + j];
        Wqf[h * (H + 1) + j] = s;
    }
}

// ---------------------------------------------------------------------------
// WMMA GEMM (NT):  C[m,n] = sum_k A[m,k] * W[n,k]   K=128, N=128 (8 tiles), M mult of 16
// One wave (32 threads) per 16x16 output tile.
// ---------------------------------------------------------------------------
__global__ __launch_bounds__(32)
void k_gemm_nt_wmma(const float* __restrict__ A, int lda,
                    const float* __restrict__ W, int ldw,
                    float* __restrict__ C, int ldc, int K) {
    const int mt   = blockIdx.x * 16;
    const int nt   = blockIdx.y * 16;
    const int lane = threadIdx.x & 31;
    const int row  = lane & 15;
    const int half = lane >> 4;

    v8f c = {};

#if __has_builtin(__builtin_amdgcn_wmma_f32_16x16x4_f32)
    // f32 WMMA, full precision.
    // A frag (16x4): lane<16 holds M=row {K0,K1}; lane>=16 holds M=row {K2,K3}.
    // B frag (4x16): lane<16 holds N=row {K0,K1}; lane>=16 holds N=row {K2,K3}.
    const float* Ar = A + (size_t)(mt + row) * lda + 2 * half;
    const float* Wr = W + (size_t)(nt + row) * ldw + 2 * half;
    for (int k = 0; k < K; k += 4) {
        v2f a  = { Ar[k], Ar[k + 1] };
        v2f bb = { Wr[k], Wr[k + 1] };
        c = __builtin_amdgcn_wmma_f32_16x16x4_f32(false, a, false, bb,
                                                  (short)0, c, false, false);
    }
#else
    // Fallback: probe-verified f16 16x16x32 path (convert on the fly).
    const float* Ar = A + (size_t)(mt + row) * lda;
    const float* Wr = W + (size_t)(nt + row) * ldw;
    const int h8 = half * 8;
    for (int k = 0; k < K; k += 32) {
        v16h a, b;
        #pragma unroll
        for (int e = 0; e < 16; ++e) {
            int kk = k + h8 + (e & 7) + ((e >= 8) ? 16 : 0);
            a[e] = (_Float16)Ar[kk];
            b[e] = (_Float16)Wr[kk];
        }
        c = __builtin_amdgcn_wmma_f32_16x16x32_f16(false, a, false, b,
                                                   (short)0, c, false, false);
    }
#endif

    // C/D layout: VGPR r, lanes 0-15 -> M=r, N=lane; lanes 16-31 -> M=r+8, N=lane-16.
    #pragma unroll
    for (int r = 0; r < 8; ++r)
        C[(size_t)(mt + r + 8 * half) * ldc + nt + row] = c[r];
}

// ---------------------------------------------------------------------------
// Wave32 reductions
// ---------------------------------------------------------------------------
__device__ __forceinline__ float wred_max(float v) {
    for (int m = 16; m; m >>= 1) v = fmaxf(v, __shfl_xor(v, m, 32));
    return v;
}
__device__ __forceinline__ float wred_sum(float v) {
    for (int m = 16; m; m >>= 1) v += __shfl_xor(v, m, 32);
    return v;
}

// ---------------------------------------------------------------------------
// Decode kernel: one block per batch item, 256 threads (8 wave32s).
// K, V, K2o staged in LDS once; 128 sequential steps run entirely out of LDS.
// ---------------------------------------------------------------------------
__global__ __launch_bounds__(256)
void k_decode(const float* __restrict__ Kh,  const float* __restrict__ Vm,
              const float* __restrict__ K2o, const float* __restrict__ Qe,
              const float* __restrict__ poolq, const float* __restrict__ Wqf,
              const float* __restrict__ capacity, const float* __restrict__ demand,
              const int* __restrict__ p_nsteps, const int* __restrict__ p_T,
              float* __restrict__ out, int B, int N, int H) {
    extern __shared__ float smem[];
    const int tid = threadIdx.x;
    const int b   = blockIdx.x;
    const int NH  = N * H;
    const int SS  = N + 12; // padded score stride

    float* sK    = smem;
    float* sV    = sK  + NH;
    float* sK2   = sV  + NH;
    float* sSc   = sK2 + NH;        // 8 * SS
    float* sQ    = sSc + 8 * SS;    // H
    float* sGa   = sQ  + H;         // H
    float* sGb   = sGa + H;         // H
    float* sU    = sGb + H;         // H
    float* sUp   = sU  + H;         // 2H
    float* sDem  = sUp + 2 * H;     // H
    float* sMask = sDem + H;        // H
    float* sM1   = sMask + H;       // H
    float* sQc   = sM1 + H;         // H
    float* sPq   = sQc + H;         // H
    float* sRed  = sPq + H;         // 8
    float* fM    = sRed + 8;        // 8 scalars: [0]=cap, [1]=logp_step
    int*   iM    = (int*)(fM + 8);  // 8 ints:    [0]=idx, [1]=go_depot

    // ---- stage K / V / K2o into LDS (float4) ----
    {
        const int n4 = NH / 4;
        const float4* gK = (const float4*)(Kh  + (size_t)b * NH);
        const float4* gV = (const float4*)(Vm  + (size_t)b * NH);
        const float4* g2 = (const float4*)(K2o + (size_t)b * NH);
        float4* lK = (float4*)sK;  float4* lV = (float4*)sV;  float4* l2 = (float4*)sK2;
        for (int i = tid; i < n4; i += blockDim.x) { lK[i] = gK[i]; lV[i] = gV[i]; l2[i] = g2[i]; }
    }
    if (tid < H) {
        sQc[tid]  = Wqf[tid * (H + 1) + H];           // capacity column of folded weight
        sPq[tid]  = poolq[(size_t)b * H + tid];
        sDem[tid] = (tid < N) ? demand[(size_t)b * N + tid] : 0.f;
        sM1[tid]  = 0.f;
        sMask[tid] = 0.f;
    }
    if (tid == 0) { fM[0] = capacity[b]; iM[0] = 0; iM[1] = 1; }
    __syncthreads();

    // ---- initial mask (update_mask with index=0, mask1=0) ----
    {
        float c0 = fM[0];
        if (tid < N) sMask[tid] = (sDem[tid] > c0) ? 1.f : 0.f;
    }
    __syncthreads();
    if (tid < 32) {
        int all = 1;
        for (int n = 1 + tid; n < N; n += 32) if (sMask[n] <= 0.f) all = 0;
        for (int m = 16; m; m >>= 1) all &= __shfl_xor(all, m, 32);
        if (tid == 0) sMask[0] = all ? 0.f : 1.f;   // go_depot=1 initially
    }

    const int   n_steps  = *p_nsteps;
    const float invT     = 1.0f / (float)(*p_T);
    const float base_cap = capacity[0];
    const float inv_sqrt_h = rsqrtf((float)H);

    float logp_acc = 0.f;   // thread 0 only
    int   visited  = 0;     // thread 0 only

    for (int step = 0; step < n_steps; ++step) {
        __syncthreads();
        // 1) q = Qe[b,idx] + cap * Wqf[:,H] + pool_q[b]
        {
            int idx = iM[0];
            float capv = fM[0];
            if (tid < H)
                sQ[tid] = Qe[((size_t)b * N + idx) * H + tid] + capv * sQc[tid] + sPq[tid];
        }
        __syncthreads();

        // 2) scores[h,n] = (q_h . K[n,h,:]) / 4, masked
        for (int p = tid; p < 8 * N; p += blockDim.x) {
            int n = p >> 3, h = p & 7;
            const float* kp = sK + n * H + h * 16;
            const float* qp = sQ + h * 16;
            float acc = 0.f;
            #pragma unroll
            for (int d = 0; d < 16; ++d) acc += qp[d] * kp[d];
            acc *= 0.25f;
            if (sMask[n] > 0.f) acc = NEGV;
            sSc[h * SS + n] = acc;
        }
        __syncthreads();

        // 3) softmax per head (wave w handles head w)
        {
            int h = tid >> 5, l = tid & 31;
            float mx = -3.4e38f;
            for (int n = l; n < N; n += 32) mx = fmaxf(mx, sSc[h * SS + n]);
            mx = wred_max(mx);
            float s = 0.f;
            for (int n = l; n < N; n += 32) {
                float e = __expf(sSc[h * SS + n] - mx);
                sSc[h * SS + n] = e;
                s += e;
            }
            s = wred_sum(s);
            if (l == 0) sRed[h] = 1.0f / s;
        }
        __syncthreads();

        // 4) glimpse g[hd] = sum_n attn[h,n] * V[n,hd]  (split n-range in two)
        {
            int t = tid & (H - 1);
            int part = tid >> 7;
            int h = t >> 4;
            int n0 = part ? (N / 2) : 0;
            int n1 = part ? N : (N / 2);
            float acc = 0.f;
            for (int n = n0; n < n1; ++n) acc += sSc[h * SS + n] * sV[n * H + t];
            (part ? sGb : sGa)[t] = acc;
        }
        __syncthreads();
        if (tid < H) {
            int h = tid >> 4;
            sGa[tid] = (sGa[tid] + sGb[tid]) * sRed[h];
        }
        __syncthreads();

        // 5) u[n] = 10 * tanh(g . K2o[n] / sqrt(H)), masked (2 threads per n)
        if (tid < 2 * N) {
            int n = tid >> 1, hf = tid & 1;
            const float* kp = sK2 + n * H + hf * (H / 2);
            const float* gp = sGa + hf * (H / 2);
            float acc = 0.f;
            for (int j = 0; j < H / 2; ++j) acc += gp[j] * kp[j];
            sUp[tid] = acc;
        }
        __syncthreads();
        if (tid < H) {
            float u;
            if (tid < N) {
                float v = (sUp[2 * tid] + sUp[2 * tid + 1]) * inv_sqrt_h;
                u = 10.0f * tanhf(v);
                if (sMask[tid] > 0.f) u = NEGV;
            } else u = NEGV;
            sU[tid] = u;
        }
        __syncthreads();

        // 6) argmax (first-index tie-break, matching jnp.argmax) + log-softmax at argmax
        if (tid < 32) {
            float bv = -3.4e38f; int bi = 0;
            for (int n = tid; n < N; n += 32) {
                float v = sU[n];
                if (v > bv) { bv = v; bi = n; }
            }
            for (int m = 16; m; m >>= 1) {
                float ov = __shfl_xor(bv, m, 32);
                int   oi = __shfl_xor(bi, m, 32);
                if (ov > bv || (ov == bv && oi < bi)) { bv = ov; bi = oi; }
            }
            float se = 0.f;
            for (int n = tid; n < N; n += 32) se += __expf((sU[n] - bv) * invT);
            se = wred_sum(se);
            if (tid == 0) { iM[0] = bi; fM[1] = -__logf(se); }
        }
        __syncthreads();

        // 7) state update
        if (tid == 0) {
            int i = iM[0];
            float isdone = (visited >= N - 1) ? 1.f : 0.f;
            logp_acc += fM[1] * (1.f - isdone);
            out[(size_t)b * n_steps + step] = (float)i;
            if (i != 0 && sM1[i] == 0.f) { sM1[i] = 1.f; ++visited; }
            fM[0] = (i == 0) ? base_cap : (fM[0] - sDem[i]);
            iM[1] = (i == 0) ? 1 : 0;
        }
        __syncthreads();
        {
            float c2 = fM[0];
            if (tid < N) sMask[tid] = fmaxf(sM1[tid], (sDem[tid] > c2) ? 1.f : 0.f);
        }
        __syncthreads();
        if (tid < 32) {
            int all = 1;
            for (int n = 1 + tid; n < N; n += 32) if (sMask[n] <= 0.f) all = 0;
            for (int m = 16; m; m >>= 1) all &= __shfl_xor(all, m, 32);
            if (tid == 0) sMask[0] = all ? 0.f : (iM[1] ? 1.f : 0.f);
        }
    }
    __syncthreads();
    if (tid == 0) out[(size_t)B * n_steps + b] = logp_acc;
}

// ---------------------------------------------------------------------------
// Host launcher
// ---------------------------------------------------------------------------
extern "C" void kernel_launch(void* const* d_in, const int* in_sizes, int n_in,
                              void* d_out, int out_size, void* d_ws, size_t ws_size,
                              hipStream_t stream) {
    const float* E    = (const float*)d_in[0];
    const float* pool = (const float*)d_in[1];
    const float* cap  = (const float*)d_in[2];
    const float* dem  = (const float*)d_in[3];
    const float* fc_w = (const float*)d_in[4];
    const float* fc1  = (const float*)d_in[5];
    const float* Wq   = (const float*)d_in[6];
    const float* Wk   = (const float*)d_in[7];
    const float* Wv   = (const float*)d_in[8];
    const float* Wo   = (const float*)d_in[9];
    const float* Wk2  = (const float*)d_in[10];
    const int* p_nsteps = (const int*)d_in[11];
    const int* p_T      = (const int*)d_in[12];

    const int B = in_sizes[2];                 // capacity is (B,1)
    const int H = in_sizes[1] / B;             // pool is (B,H)
    const int N = in_sizes[0] / (B * H);       // encoder_inputs is (B,N,H)

    const size_t NHt = (size_t)B * N * H;
    const size_t BH  = (size_t)B * H;
    float* ws    = (float*)d_ws;
    float* Kh    = ws;                 // B*N*H
    float* Vm    = Kh  + NHt;          // B*N*H
    float* K2o   = Vm  + NHt;          // B*N*H
    float* Qe    = K2o + NHt;          // B*N*H
    float* P1    = Qe  + NHt;          // B*H
    float* poolq = P1  + BH;           // B*H
    float* Wct   = poolq + BH;         // H*H
    float* Wqf   = Wct + (size_t)H * H;// H*(H+1)

    // Folded weights
    k_wct<<<H, H, 0, stream>>>(Wo, Wk2, Wct, H);
    k_wqf<<<H, ((H + 1 + 31) / 32) * 32, 0, stream>>>(Wq, fc_w, Wqf, H);

    // Big WMMA GEMMs: Kh = E@Wk^T, V = E@Wv^T, K2o = E@Wct^T(=E@Wk2^T@Wo), Qe = E@Wqf^T
    dim3 gbig((B * N) / 16, H / 16);
    k_gemm_nt_wmma<<<gbig, 32, 0, stream>>>(E, H, Wk,  H,     Kh,  H, H);
    k_gemm_nt_wmma<<<gbig, 32, 0, stream>>>(E, H, Wv,  H,     Vm,  H, H);
    k_gemm_nt_wmma<<<gbig, 32, 0, stream>>>(E, H, Wct, H,     K2o, H, H);
    k_gemm_nt_wmma<<<gbig, 32, 0, stream>>>(E, H, Wqf, H + 1, Qe,  H, H);

    // pool_q = pool @ fc1^T @ Wq^T
    dim3 gsm(B / 16, H / 16);
    k_gemm_nt_wmma<<<gsm, 32, 0, stream>>>(pool, H, fc1, H, P1,    H, H);
    k_gemm_nt_wmma<<<gsm, 32, 0, stream>>>(P1,   H, Wq,  H, poolq, H, H);

    // Decode: one block per batch item; K/V/K2o staged in dynamic LDS (~163 KB, <320 KB/WG).
    const int SS = N + 12;
    const size_t shbytes = (size_t)(3 * N * H + 8 * SS + 11 * H + 24) * sizeof(float);
    k_decode<<<B, 256, shbytes, stream>>>(Kh, Vm, K2o, Qe, poolq, Wqf, cap, dem,
                                          p_nsteps, p_T, (float*)d_out, B, N, H);
}
